// MultiHeadSelfAttention_72267119722872
// MI455X (gfx1250) — matrix operand order repaired and checked by reading, not compiled, not run on hardware
//
#include <hip/hip_runtime.h>
#include <math.h>

// Problem dims (match reference)
#define B_  4
#define S_  2048
#define D_  1024
#define H_  16
#define DK_ 64

typedef __attribute__((ext_vector_type(16))) __bf16 v16bf;
typedef __attribute__((ext_vector_type(8)))  __bf16 v8bf;
typedef __attribute__((ext_vector_type(8)))  float  v8f;
typedef __attribute__((ext_vector_type(4)))  unsigned int v4u;
typedef __attribute__((ext_vector_type(8)))  int v8i_;
typedef __attribute__((ext_vector_type(4)))  int v4i_;

static __device__ __forceinline__ unsigned short f2bf(float f) {
  unsigned int u = __float_as_uint(f);
  u += 0x7FFFu + ((u >> 16) & 1u);   // round-to-nearest-even
  return (unsigned short)(u >> 16);
}

// Build a 16x32 bf16 A-fragment: lane holds row lane%16, elements 0..15 at
// k0..k0+15 (k0 = (lane/16)*16). Caller passes ptr to k0 chunk.
static __device__ __forceinline__ v16bf make_afrag(const unsigned short* p) {
  v8bf lo = *(const v8bf*)(p);
  v8bf hi = *(const v8bf*)(p + 8);
  return __builtin_shufflevector(lo, hi, 0,1,2,3,4,5,6,7,8,9,10,11,12,13,14,15);
}

static __device__ __forceinline__ v8f wmma_bf16(v16bf a, v16bf b, v8f c) {
  return __builtin_amdgcn_wmma_f32_16x16x32_bf16(false, a, false, b,
                                                 (short)0, c, false, false);
}

// Issue one TDM 2-D tile load (bf16 elements) global -> LDS.
// Descriptor bit layout per CDNA5 ISA.
// tile_d1 == 0 => 1-D tile. stride0 in elements. Tracked by TENSORcnt.
// amdgpu-toolchain (clang-23) builtin arity: 6 args
//   (uint32x4 g0, int32x8 g1, int32x4 g2, int32x4 g3, int32x8 extra, i32 cpol)
static __device__ __forceinline__ void tdm_load_tile(
    unsigned int lds_byte_off, const unsigned short* gsrc,
    unsigned int tile_d0, unsigned int tile_d1,
    unsigned int tensor_d0, unsigned int tensor_d1,
    unsigned long long stride0)
{
  unsigned long long ga = (unsigned long long)(uintptr_t)gsrc;
  v4u g0;
  g0[0] = 1u;                                   // count=1, user descriptor
  g0[1] = lds_byte_off;                         // lds_addr (bytes)
  g0[2] = (unsigned int)(ga & 0xFFFFFFFFu);     // global_addr[31:0]
  g0[3] = (unsigned int)((ga >> 32) & 0x01FFFFFFu) | 0x80000000u; // addr[56:32] | type=2
  v8i_ g1;
  g1[0] = (int)(1u << 16);                      // workgroup_mask=0, data_size=2B
  g1[1] = (int)((tensor_d0 & 0xFFFFu) << 16);   // tensor_dim0[15:0] @ bits63:48
  g1[2] = (int)((tensor_d0 >> 16) | ((tensor_d1 & 0xFFFFu) << 16));
  g1[3] = (int)((tensor_d1 >> 16) | (tile_d0 << 16));   // tile_dim0 @ bits127:112
  g1[4] = (int)(tile_d1 & 0xFFFFu);             // tile_dim1 (0 => 1-D), tile_dim2=0
  g1[5] = (int)(unsigned int)(stride0 & 0xFFFFFFFFull); // tensor_dim0_stride[31:0]
  g1[6] = (int)(unsigned int)((stride0 >> 32) & 0xFFFFull);
  g1[7] = 0;
  v4i_ gz4 = {0, 0, 0, 0};                      // groups 2/3 unused (<=2-D)
  v8i_ gz8 = {0, 0, 0, 0, 0, 0, 0, 0};
  __builtin_amdgcn_tensor_load_to_lds(g0, g1, gz4, gz4, gz8, 0);
}

// ---------------------------------------------------------------- convert
__global__ void cvt_bf16_kernel(const float4* __restrict__ in,
                                ushort4* __restrict__ out, int n4) {
  int i = blockIdx.x * blockDim.x + threadIdx.x;
  int stride = gridDim.x * blockDim.x;
  for (; i < n4; i += stride) {
    float4 f = in[i];
    ushort4 o;
    o.x = f2bf(f.x); o.y = f2bf(f.y); o.z = f2bf(f.z); o.w = f2bf(f.w);
    out[i] = o;
  }
}

// ------------------------------------------------- fused QKV GEMM + RoPE
// out[m,o] = sum_d x[m,d] * W[o,d]   (X * W^T), M=8192, N=K=1024
// blockIdx.z selects Q(0)/K(1)/V(2). 8 waves: 4(M) x 2(N); wave tile 32x64.
// Software-pipelined; unroll-2 lets regalloc ping-pong instead of copying.
__global__ __launch_bounds__(256) void qkv_rope_gemm(
    const unsigned short* __restrict__ xb,
    const unsigned short* __restrict__ wqb,
    const unsigned short* __restrict__ wkb,
    const unsigned short* __restrict__ wvb,
    unsigned short* __restrict__ Qb,   // [B,H,S,DK] bf16, RoPE applied
    unsigned short* __restrict__ Kb,   // [B,H,S,DK] bf16, RoPE applied
    unsigned short* __restrict__ Vt)   // [B,H,DK,S] bf16 (transposed)
{
  const int z = blockIdx.z;
  const unsigned short* W = (z == 0) ? wqb : (z == 1) ? wkb : wvb;
  const int lane = threadIdx.x & 31;
  const int wave = threadIdx.x >> 5;
  const int half = lane >> 4;
  const int col  = lane & 15;
  const int m0 = blockIdx.x * 128 + (wave >> 1) * 32;
  const int n0 = blockIdx.y * 128 + (wave & 1) * 64;

  const v8f vzero = {0.f,0.f,0.f,0.f,0.f,0.f,0.f,0.f};
  v8f acc0[4], acc1[4];
  #pragma unroll
  for (int j = 0; j < 4; ++j) { acc0[j] = vzero; acc1[j] = vzero; }

  const unsigned short* arow0 = xb + (m0 + col) * D_ + half * 16;
  const unsigned short* arow1 = arow0 + 16 * D_;
  const unsigned short* brow  = W + (n0 + col) * D_ + half * 16;

  // prologue: fragments for kb = 0
  v16bf a0 = make_afrag(arow0);
  v16bf a1 = make_afrag(arow1);
  v16bf bf[4];
  #pragma unroll
  for (int j = 0; j < 4; ++j) bf[j] = *(const v16bf*)(brow + j * 16 * D_);

  #pragma unroll 2
  for (int kb = 0; kb < D_; kb += 32) {
    const int kbn = (kb + 32 < D_) ? kb + 32 : 0;   // wrap: harmless reload
    // issue next step's loads first (stay outstanding across the WMMAs)
    v16bf na0 = make_afrag(arow0 + kbn);
    v16bf na1 = make_afrag(arow1 + kbn);
    v16bf nb[4];
    #pragma unroll
    for (int j = 0; j < 4; ++j) nb[j] = *(const v16bf*)(brow + j * 16 * D_ + kbn);
    #pragma unroll
    for (int j = 0; j < 4; ++j) acc0[j] = wmma_bf16(a0, bf[j], acc0[j]);
    #pragma unroll
    for (int j = 0; j < 4; ++j) acc1[j] = wmma_bf16(a1, bf[j], acc1[j]);
    a0 = na0; a1 = na1;
    #pragma unroll
    for (int j = 0; j < 4; ++j) bf[j] = nb[j];
  }

  // Epilogue: C-frag element v -> (row base + half*8 + v, col n0+j*16+col)
  #pragma unroll
  for (int r = 0; r < 2; ++r) {
    #pragma unroll
    for (int j = 0; j < 4; ++j) {
      const int o = n0 + j*16 + col;
      const int h = o >> 6;
      const int d = o & (DK_ - 1);
      // inv_freq = THETA^(-(2*(d/2))/DK) ; ln(10000)=9.210340372
      const float invf = __expf(-9.210340371976184f * (float)(d & ~1) * (1.0f/64.0f));
      #pragma unroll
      for (int v = 0; v < 8; ++v) {
        const int m = m0 + r*16 + half*8 + v;
        const int b = m >> 11;          // / S_
        const int s = m & (S_ - 1);
        float val = (r == 0) ? acc0[j][v] : acc1[j][v];
        if (z == 2) {
          Vt[((b*H_ + h) * DK_ + d) * S_ + s] = f2bf(val);
        } else {
          // RoPE: even/odd features live in adjacent lanes of the C-frag
          float oth = __shfl_xor(val, 1);
          float rot = (o & 1) ? oth : -oth;     // even: -x[o+1], odd: +x[o-1]
          float ang = (float)s * invf;
          float sn, cs;
          __sincosf(ang, &sn, &cs);
          float rr = val * cs + rot * sn;
          unsigned short* dst = (z == 0) ? Qb : Kb;
          dst[((b*H_ + h) * S_ + s) * DK_ + d] = f2bf(rr);
        }
      }
    }
  }
}

// ------------------------------------------------------ flash attention
// grid: (S/64, B*H), 128 threads = 4 waves; each wave owns a 16-row q tile.
// K/V tiles for each 32-key block are DMA'd into double-buffered LDS by the
// Tensor Data Mover (wave 0 issues tensor_load_to_lds; s_wait_tensorcnt +
// one barrier per block pipelines DMA of block i+1 under compute of block i).
__global__ __launch_bounds__(128) void flash_attn(
    const unsigned short* __restrict__ Qb,
    const unsigned short* __restrict__ Kb,
    const unsigned short* __restrict__ Vt,
    unsigned short* __restrict__ ctx)   // [B,S,D] bf16
{
  __shared__ alignas(64) unsigned short ldsP[4][16*32];   //  4 KB P transpose
  __shared__ alignas(64) unsigned short Kbuf[2][32*64];   //  8 KB K tiles
  __shared__ alignas(64) unsigned short Vbuf[2][64*32];   //  8 KB V tiles
  const int lane = threadIdx.x & 31;
  const int wave = threadIdx.x >> 5;
  const int half = lane >> 4;
  const int col  = lane & 15;
  const int bh = blockIdx.y;
  const int b  = bh >> 4;
  const int h  = bh & 15;
  const int qblk0 = blockIdx.x * 64;
  const int q0 = qblk0 + wave * 16;

  const unsigned short* Qh = Qb + (size_t)bh * S_ * DK_;
  const unsigned short* Kh = Kb + (size_t)bh * S_ * DK_;
  const unsigned short* Vh = Vt + (size_t)bh * DK_ * S_;

  v16bf qf[2];
  #pragma unroll
  for (int c = 0; c < 2; ++c)
    qf[c] = make_afrag(Qh + (q0 + col) * DK_ + c*32 + half*16);

  const v8f vzero = {0.f,0.f,0.f,0.f,0.f,0.f,0.f,0.f};
  v8f oacc[4];
  #pragma unroll
  for (int t = 0; t < 4; ++t) oacc[t] = vzero;
  float mrow[8], lrow[8];
  #pragma unroll
  for (int v = 0; v < 8; ++v) { mrow[v] = -__builtin_inff(); lrow[v] = 0.0f; }

  const int nkb_blk = qblk0 / 32 + 2;        // uniform across the block
  const int nkb_w   = (q0 + 15) / 32 + 1;    // this wave's causal limit

  if (wave == 0) {
    // prologue: DMA key block 0 into buffer 0
    tdm_load_tile((unsigned int)(uintptr_t)&Kbuf[0][0], Kh, /*tile*/2048, 0,
                  /*tensor*/2048, 1, 2048);                    // 32x64 contiguous
    tdm_load_tile((unsigned int)(uintptr_t)&Vbuf[0][0], Vh, /*tile*/32, 64,
                  /*tensor*/S_, DK_, /*stride0*/S_);           // 64 rows x 32 cols
  }

  for (int kbi = 0; kbi < nkb_blk; ++kbi) {
    const int kb  = kbi * 32;
    const int cur = kbi & 1;
    if (wave == 0) __builtin_amdgcn_s_wait_tensorcnt(0);
    __syncthreads();                       // buf[cur] ready; buf[cur^1] free
    if (wave == 0 && kbi + 1 < nkb_blk) {  // DMA next block under this compute
      const int kbn = kb + 32;
      tdm_load_tile((unsigned int)(uintptr_t)&Kbuf[cur ^ 1][0],
                    Kh + (size_t)kbn * DK_, 2048, 0, 2048, 1, 2048);
      tdm_load_tile((unsigned int)(uintptr_t)&Vbuf[cur ^ 1][0],
                    Vh + kbn, 32, 64, S_, DK_, S_);
    }
    if (kbi < nkb_w) {
      // K/V fragments straight from LDS (ds_load_b128 pairs)
      v16bf kf[2][2];
      #pragma unroll
      for (int c = 0; c < 2; ++c) {
        kf[0][c] = *(const v16bf*)(&Kbuf[cur][(     col) * DK_ + c*32 + half*16]);
        kf[1][c] = *(const v16bf*)(&Kbuf[cur][(16 + col) * DK_ + c*32 + half*16]);
      }
      v16bf vf[4];
      #pragma unroll
      for (int t = 0; t < 4; ++t)
        vf[t] = *(const v16bf*)(&Vbuf[cur][(t*16 + col) * 32 + half*16]);

      v8f sc0 = vzero, sc1 = vzero;
      #pragma unroll
      for (int c = 0; c < 2; ++c) {
        sc0 = wmma_bf16(qf[c], kf[0][c], sc0);
        sc1 = wmma_bf16(qf[c], kf[1][c], sc1);
      }
      const bool need_mask = (kb + 31 > q0);
      float s0[8], s1[8];
      #pragma unroll
      for (int v = 0; v < 8; ++v) {
        const int srow = q0 + half*8 + v;
        float a0 = sc0[v] * 0.125f;        // 1/sqrt(DK)
        float a1 = sc1[v] * 0.125f;
        if (need_mask) {
          if (kb +      col > srow) a0 = -__builtin_inff();
          if (kb + 16 + col > srow) a1 = -__builtin_inff();
        }
        s0[v] = a0; s1[v] = a1;
      }
      float corr[8];
      #pragma unroll
      for (int v = 0; v < 8; ++v) {
        float mx = fmaxf(s0[v], s1[v]);
        mx = fmaxf(mx, __shfl_xor(mx, 1));
        mx = fmaxf(mx, __shfl_xor(mx, 2));
        mx = fmaxf(mx, __shfl_xor(mx, 4));
        mx = fmaxf(mx, __shfl_xor(mx, 8));
        float mnew = fmaxf(mrow[v], mx);
        float p0 = __expf(s0[v] - mnew);
        float p1 = __expf(s1[v] - mnew);
        float ps = p0 + p1;
        ps += __shfl_xor(ps, 1);
        ps += __shfl_xor(ps, 2);
        ps += __shfl_xor(ps, 4);
        ps += __shfl_xor(ps, 8);
        float c_ = __expf(mrow[v] - mnew);
        lrow[v] = lrow[v] * c_ + ps;
        mrow[v] = mnew;
        corr[v] = c_;
        ldsP[wave][(half*8 + v)*32 +      col] = f2bf(p0);
        ldsP[wave][(half*8 + v)*32 + 16 + col] = f2bf(p1);
      }
      #pragma unroll
      for (int t = 0; t < 4; ++t)
        #pragma unroll
        for (int v = 0; v < 8; ++v)
          oacc[t][v] *= corr[v];
      // LDS is in-order per wave; fence stops compiler reordering store->load
      asm volatile("s_wait_dscnt 0" ::: "memory");
      v16bf pf = make_afrag(&ldsP[wave][col*32 + half*16]);  // C- -> A-layout
      #pragma unroll
      for (int t = 0; t < 4; ++t)
        oacc[t] = wmma_bf16(pf, vf[t], oacc[t]);
      asm volatile("s_wait_dscnt 0" ::: "memory");
    }
  }

  float linv[8];
  #pragma unroll
  for (int v = 0; v < 8; ++v) linv[v] = 1.0f / lrow[v];
  #pragma unroll
  for (int t = 0; t < 4; ++t) {
    #pragma unroll
    for (int v = 0; v < 8; ++v) {
      const int s = q0 + half*8 + v;
      const int o = h*DK_ + t*16 + col;
      ctx[(size_t)(b * S_ + s) * D_ + o] = f2bf(oacc[t][v] * linv[v]);
    }
  }
}

// ------------------------------------------------- output projection (fp32 out)
__global__ __launch_bounds__(256) void out_proj_gemm(
    const unsigned short* __restrict__ ctx,
    const unsigned short* __restrict__ wob,
    float* __restrict__ out)
{
  const int lane = threadIdx.x & 31;
  const int wave = threadIdx.x >> 5;
  const int half = lane >> 4;
  const int col  = lane & 15;
  const int m0 = blockIdx.x * 128 + (wave >> 1) * 32;
  const int n0 = blockIdx.y * 128 + (wave & 1) * 64;

  const v8f vzero = {0.f,0.f,0.f,0.f,0.f,0.f,0.f,0.f};
  v8f acc0[4], acc1[4];
  #pragma unroll
  for (int j = 0; j < 4; ++j) { acc0[j] = vzero; acc1[j] = vzero; }

  const unsigned short* arow0 = ctx + (m0 + col) * D_ + half * 16;
  const unsigned short* arow1 = arow0 + 16 * D_;
  const unsigned short* brow  = wob + (n0 + col) * D_ + half * 16;

  v16bf a0 = make_afrag(arow0);
  v16bf a1 = make_afrag(arow1);
  v16bf bf[4];
  #pragma unroll
  for (int j = 0; j < 4; ++j) bf[j] = *(const v16bf*)(brow + j * 16 * D_);

  #pragma unroll 2
  for (int kb = 0; kb < D_; kb += 32) {
    const int kbn = (kb + 32 < D_) ? kb + 32 : 0;
    v16bf na0 = make_afrag(arow0 + kbn);
    v16bf na1 = make_afrag(arow1 + kbn);
    v16bf nb[4];
    #pragma unroll
    for (int j = 0; j < 4; ++j) nb[j] = *(const v16bf*)(brow + j * 16 * D_ + kbn);
    #pragma unroll
    for (int j = 0; j < 4; ++j) acc0[j] = wmma_bf16(a0, bf[j], acc0[j]);
    #pragma unroll
    for (int j = 0; j < 4; ++j) acc1[j] = wmma_bf16(a1, bf[j], acc1[j]);
    a0 = na0; a1 = na1;
    #pragma unroll
    for (int j = 0; j < 4; ++j) bf[j] = nb[j];
  }

  #pragma unroll
  for (int j = 0; j < 4; ++j)
    #pragma unroll
    for (int v = 0; v < 8; ++v) {
      out[(size_t)(m0      + half*8 + v) * D_ + n0 + j*16 + col] = acc0[j][v];
      out[(size_t)(m0 + 16 + half*8 + v) * D_ + n0 + j*16 + col] = acc1[j][v];
    }
}

// ---------------------------------------------------------------- launch
extern "C" void kernel_launch(void* const* d_in, const int* in_sizes, int n_in,
                              void* d_out, int out_size, void* d_ws, size_t ws_size,
                              hipStream_t stream) {
  (void)in_sizes; (void)n_in; (void)out_size; (void)ws_size;
  const float* x  = (const float*)d_in[0];
  const float* wq = (const float*)d_in[1];
  const float* wk = (const float*)d_in[2];
  const float* wv = (const float*)d_in[3];
  const float* wo = (const float*)d_in[4];
  float* out = (float*)d_out;

  // Workspace layout (bytes). Total = 75,497,472.
  char* ws = (char*)d_ws;
  unsigned short* xb  = (unsigned short*)(ws + 0);          // 16 MB  (reused as ctx)
  unsigned short* wqb = (unsigned short*)(ws + 16777216);   //  2 MB
  unsigned short* wkb = (unsigned short*)(ws + 18874368);   //  2 MB
  unsigned short* wvb = (unsigned short*)(ws + 20971520);   //  2 MB
  unsigned short* wob = (unsigned short*)(ws + 23068672);   //  2 MB
  unsigned short* Qb  = (unsigned short*)(ws + 25165824);   // 16 MB
  unsigned short* Kb  = (unsigned short*)(ws + 41943040);   // 16 MB
  unsigned short* Vt  = (unsigned short*)(ws + 58720256);   // 16 MB
  unsigned short* ctx = xb;  // x is dead after the QKV GEMM

  const int nx4 = (B_ * S_ * D_) / 4;
  const int nw4 = (D_ * D_) / 4;
  cvt_bf16_kernel<<<1024, 256, 0, stream>>>((const float4*)x,  (ushort4*)xb,  nx4);
  cvt_bf16_kernel<<<512,  256, 0, stream>>>((const float4*)wq, (ushort4*)wqb, nw4);
  cvt_bf16_kernel<<<512,  256, 0, stream>>>((const float4*)wk, (ushort4*)wkb, nw4);
  cvt_bf16_kernel<<<512,  256, 0, stream>>>((const float4*)wv, (ushort4*)wvb, nw4);
  cvt_bf16_kernel<<<512,  256, 0, stream>>>((const float4*)wo, (ushort4*)wob, nw4);

  qkv_rope_gemm<<<dim3(64, 8, 3), 256, 0, stream>>>(xb, wqb, wkb, wvb, Qb, Kb, Vt);
  flash_attn<<<dim3(S_ / 64, B_ * H_), 128, 0, stream>>>(Qb, Kb, Vt, ctx);
  out_proj_gemm<<<dim3(64, 8), 256, 0, stream>>>(ctx, wob, out);
}
